// MultiHeadAttention_61942018343045
// MI455X (gfx1250) — matrix operand order repaired and checked by reading, not compiled, not run on hardware
//
#include <hip/hip_runtime.h>
#include <hip/hip_bf16.h>

// MHA forward for MI455X (gfx1250, wave32, WMMA).
// B=4, S=2048, D=1024, H=16, DH=64.
// All GEMMs run on v_wmma_f32_16x16x32_bf16 (bf16 A/B, f32 accum).
// bf16-only tile copies use global_load_async_to_lds_b128 (ASYNCcnt path).
// f32->bf16 staging uses v_cvt_pk_bf16_f32 (1 VALU op per pair, RNE).

#define Bc 4
#define Sc 2048
#define Dc 1024
#define Hc 16
#define DHc 64

typedef __attribute__((ext_vector_type(16))) __bf16        v16bf;
typedef __attribute__((ext_vector_type(2)))  __bf16        v2bf;
typedef __attribute__((ext_vector_type(8)))  float         v8f;
typedef __attribute__((ext_vector_type(8)))  unsigned int  v8u;

union FragU { v8u u; v16bf b; };

__device__ __forceinline__ unsigned short f32_to_bf16(float f) {
  unsigned int u = __builtin_bit_cast(unsigned int, f);
  u += 0x7fffu + ((u >> 16) & 1u);           // round to nearest even
  return (unsigned short)(u >> 16);
}

#if __has_builtin(__builtin_amdgcn_cvt_pk_bf16_f32)
__device__ __forceinline__ unsigned pack_bf16x2(float lo, float hi) {
  v2bf p = __builtin_amdgcn_cvt_pk_bf16_f32(lo, hi);   // 1 VALU op, RNE
  return __builtin_bit_cast(unsigned, p);
}
#else
// gfx1250 has V_CVT_PK_BF16_F32 but the builtin is gated to gfx950 in this
// toolchain -- use inline asm (1 VALU op per packed pair, RNE).
__device__ __forceinline__ unsigned pack_bf16x2(float lo, float hi) {
  unsigned r;
  asm("v_cvt_pk_bf16_f32 %0, %1, %2" : "=v"(r) : "v"(lo), "v"(hi));
  return r;
}
#endif

// CDNA5 async memory -> LDS copy (16B per lane), tracked by ASYNCcnt.
__device__ __forceinline__ void async_b128(unsigned lds_off, const void* gptr) {
  asm volatile("global_load_async_to_lds_b128 %0, %1, off"
               :: "v"(lds_off), "v"(gptr)
               : "memory");
}
__device__ __forceinline__ void wait_async() {
  asm volatile("s_wait_asynccnt 0x0" ::: "memory");
}

// A fragment: 16x32 bf16 tile. lane = m (0..15, both halves), K striped:
// VGPR v holds K = (v/4)*16 + (lane/16)*8 + (v%4)*2 (+k0), packed pair.
__device__ __forceinline__ v16bf load_a_frag(const unsigned short* lds, int ld,
                                             int m0, int k0) {
  const int lane = threadIdx.x & 31;
  const int m  = m0 + (lane & 15);
  const int kh = (lane >> 4) << 3;
  FragU f;
#pragma unroll
  for (int v = 0; v < 8; ++v) {
    const int k = k0 + ((v >> 2) << 4) + kh + ((v & 3) << 1);
    f.u[v] = *(const unsigned int*)(lds + m * ld + k);
  }
  return f.b;
}

// B fragment: 32x16 bf16 tile, staged transposed in LDS as [n][k] (ld = K stride).
// lane = n (0..15), VGPR v holds K = (v/4)*16 + (lane/16)*8 + (v%4)*2 (+k0), packed pair.
__device__ __forceinline__ v16bf load_b_frag(const unsigned short* ldsT, int ld,
                                             int n0, int k0) {
  const int lane = threadIdx.x & 31;
  const int n  = n0 + (lane & 15);
  const int kh = (lane >> 4) << 3;
  FragU f;
#pragma unroll
  for (int v = 0; v < 8; ++v) {
    const int k = k0 + ((v >> 2) << 4) + kh + ((v & 3) << 1);
    f.u[v] = *(const unsigned int*)(ldsT + n * ld + k);
  }
  return f.b;
}

#define WMMA_BF16(a, b, c) \
  __builtin_amdgcn_wmma_f32_16x16x32_bf16(false, (a), false, (b), (short)0, (c), false, false)

// Transposed B-stage helper: 8 consecutive f32 along n -> bf16 column stores.
__device__ __forceinline__ void stage_b_transposed(unsigned short* bp, int cb,
                                                   const float4 w0, const float4 w1) {
  const unsigned p0 = pack_bf16x2(w0.x, w0.y);
  const unsigned p1 = pack_bf16x2(w0.z, w0.w);
  const unsigned p2 = pack_bf16x2(w1.x, w1.y);
  const unsigned p3 = pack_bf16x2(w1.z, w1.w);
  bp[(cb + 0) * 32] = (unsigned short)p0;
  bp[(cb + 1) * 32] = (unsigned short)(p0 >> 16);
  bp[(cb + 2) * 32] = (unsigned short)p1;
  bp[(cb + 3) * 32] = (unsigned short)(p1 >> 16);
  bp[(cb + 4) * 32] = (unsigned short)p2;
  bp[(cb + 5) * 32] = (unsigned short)(p2 >> 16);
  bp[(cb + 6) * 32] = (unsigned short)p3;
  bp[(cb + 7) * 32] = (unsigned short)(p3 >> 16);
}

// ---------------------------------------------------------------------------
// Kernel 1: QKV projection. Out[b,h,s,dh] = X[b,s,:] @ W[:, h*64+dh] + bias.
// M=8192 (b,s), N=1024 (h,dh), K=1024. Output bf16 in [B,H,S,DH] layout.
// grid (128, 16), block 256.
// ---------------------------------------------------------------------------
__global__ void mha_qkv_proj(const float* __restrict__ X,
                             const float* __restrict__ W,
                             const float* __restrict__ bias,
                             unsigned short* __restrict__ Out) {
  __shared__ unsigned short Alds[64 * 32];   // [m][k]
  __shared__ unsigned short Blds[64 * 32];   // [n][k] (W tile transposed)
  const int m0 = blockIdx.x * 64;
  const int n0 = blockIdx.y * 64;
  const int t  = threadIdx.x;
  const int wave = t >> 5;
  const int wn = (wave & 3) << 4;            // 0,16,32,48
  const int wm = (wave >> 2) << 5;           // 0,32
  const int ra = t >> 2, ca = (t & 3) << 3;  // A stage: 8 contiguous floats
  const int rb = t >> 3, cb = (t & 7) << 3;  // B stage: 8 contiguous floats
  v8f c0 = {}; v8f c1 = {};

  for (int kk = 0; kk < Dc; kk += 32) {
    {  // stage A: f32 -> bf16, vectorized (2x float4 in, 1x uint4 out)
      const float4* src = (const float4*)(X + (size_t)(m0 + ra) * Dc + kk + ca);
      const float4 x0 = src[0], x1 = src[1];
      uint4 pk;
      pk.x = pack_bf16x2(x0.x, x0.y); pk.y = pack_bf16x2(x0.z, x0.w);
      pk.z = pack_bf16x2(x1.x, x1.y); pk.w = pack_bf16x2(x1.z, x1.w);
      *(uint4*)(Alds + ra * 32 + ca) = pk;
    }
    {  // stage B transposed: W[k][n] -> Blds[n][k], f32 -> bf16
      const float4* ws = (const float4*)(W + (size_t)(kk + rb) * 1024 + n0 + cb);
      stage_b_transposed(Blds + rb, cb, ws[0], ws[1]);
    }
    if (kk + 32 < Dc) {                      // prefetch next tiles
      __builtin_prefetch(&X[(size_t)(m0 + ra) * Dc + kk + 32], 0, 1);
      __builtin_prefetch(&W[(size_t)(kk + 32 + rb) * 1024 + n0], 0, 1);
    }
    __syncthreads();
    const v16bf a0 = load_a_frag(Alds, 32, wm, 0);
    const v16bf a1 = load_a_frag(Alds, 32, wm + 16, 0);
    const v16bf b  = load_b_frag(Blds, 32, wn, 0);
    c0 = WMMA_BF16(a0, b, c0);
    c1 = WMMA_BF16(a1, b, c1);
    __syncthreads();
  }

  const int lane = t & 31;
  const int ln = lane & 15, lh = lane >> 4;
  const int n  = n0 + wn + ln;
  const float bv = bias[n];
  const int h = n >> 6, dh = n & 63;
#pragma unroll
  for (int r = 0; r < 8; r += 2) {           // pairwise convert, scatter stores
    const unsigned p0 = pack_bf16x2(c0[r] + bv, c0[r + 1] + bv);
    const unsigned p1 = pack_bf16x2(c1[r] + bv, c1[r + 1] + bv);
    {
      const int m = m0 + wm + (lh << 3) + r;
      const int bb = m >> 11, s = m & 2047;
      unsigned short* o = Out + (((size_t)bb * Hc + h) * Sc + s) * DHc + dh;
      o[0] = (unsigned short)p0;
      o[DHc] = (unsigned short)(p0 >> 16);   // next s -> +DHc in [B,H,S,DH]
    }
    {
      const int m = m0 + wm + 16 + (lh << 3) + r;
      const int bb = m >> 11, s = m & 2047;
      unsigned short* o = Out + (((size_t)bb * Hc + h) * Sc + s) * DHc + dh;
      o[0] = (unsigned short)p1;
      o[DHc] = (unsigned short)(p1 >> 16);
    }
  }
}

// ---------------------------------------------------------------------------
// Kernel 2: scores = (Q @ K^T) / sqrt(DH), written (pre-softmax) to alpha.
// Per (b,h): M=N=2048, K=64. Q/K tiles copied with async-to-LDS (no VGPRs).
// grid (32, 32, 64), block 256.
// ---------------------------------------------------------------------------
__global__ void mha_scores(const unsigned short* __restrict__ Qh,
                           const unsigned short* __restrict__ Kh,
                           float* __restrict__ alpha) {
  __shared__ unsigned short Qlds[64 * 64];   // [q][k]
  __shared__ unsigned short Klds[64 * 64];   // [n][k] == natural Kh layout
  const int q0 = blockIdx.x * 64;
  const int n0 = blockIdx.y * 64;
  const int bh = blockIdx.z;
  const int t  = threadIdx.x;
  const int wave = t >> 5;
  const int wn = (wave & 3) << 4;
  const int wm = (wave >> 2) << 5;

  // Async copy 8KB per tile: 512 x 16B chunks, 2 per thread.
  const unsigned qlds0 = (unsigned)(size_t)(void*)Qlds;
  const unsigned klds0 = (unsigned)(size_t)(void*)Klds;
  const char* qsrc = (const char*)(Qh + ((size_t)bh * Sc + q0) * DHc);
  const char* ksrc = (const char*)(Kh + ((size_t)bh * Sc + n0) * DHc);
#pragma unroll
  for (int i = 0; i < 2; ++i) {
    const int off = (t + i * 256) * 16;
    async_b128(qlds0 + off, qsrc + off);
    async_b128(klds0 + off, ksrc + off);
  }
  wait_async();
  __syncthreads();

  v8f c0 = {}; v8f c1 = {};
#pragma unroll
  for (int kk = 0; kk < DHc; kk += 32) {
    const v16bf a0 = load_a_frag(Qlds, 64, wm, kk);
    const v16bf a1 = load_a_frag(Qlds, 64, wm + 16, kk);
    const v16bf b  = load_b_frag(Klds, 64, wn, kk);
    c0 = WMMA_BF16(a0, b, c0);
    c1 = WMMA_BF16(a1, b, c1);
  }

  const float scale = 0.125f;                // 1/sqrt(64)
  const int lane = t & 31;
  const int ln = lane & 15, lh = lane >> 4;
  const int col = n0 + wn + ln;
  float* arow = alpha + (size_t)bh * Sc * Sc + col;
#pragma unroll
  for (int r = 0; r < 8; ++r) {
    arow[(size_t)(q0 + wm + (lh << 3) + r) * Sc]      = c0[r] * scale;
    arow[(size_t)(q0 + wm + 16 + (lh << 3) + r) * Sc] = c1[r] * scale;
  }
}

// ---------------------------------------------------------------------------
// Kernel 3: in-place row softmax over alpha. One block per row of 2048.
// grid (131072), block 256.
// ---------------------------------------------------------------------------
__global__ void mha_softmax(float* __restrict__ alpha) {
  __shared__ float red[256];
  float* p = alpha + (size_t)blockIdx.x * Sc;
  const int t = threadIdx.x;
  const int base = t << 3;                   // 8 contiguous floats per thread
  float4 u0 = *(const float4*)(p + base);
  float4 u1 = *(const float4*)(p + base + 4);
  float vals[8] = {u0.x, u0.y, u0.z, u0.w, u1.x, u1.y, u1.z, u1.w};
  float m = -3.402823466e+38f;
#pragma unroll
  for (int i = 0; i < 8; ++i) m = fmaxf(m, vals[i]);
  red[t] = m; __syncthreads();
  for (int s = 128; s > 0; s >>= 1) {
    if (t < s) red[t] = fmaxf(red[t], red[t + s]);
    __syncthreads();
  }
  m = red[0]; __syncthreads();
  float sum = 0.f;
#pragma unroll
  for (int i = 0; i < 8; ++i) { vals[i] = __expf(vals[i] - m); sum += vals[i]; }
  red[t] = sum; __syncthreads();
  for (int s = 128; s > 0; s >>= 1) {
    if (t < s) red[t] += red[t + s];
    __syncthreads();
  }
  const float inv = 1.0f / red[0];
  u0.x = vals[0] * inv; u0.y = vals[1] * inv; u0.z = vals[2] * inv; u0.w = vals[3] * inv;
  u1.x = vals[4] * inv; u1.y = vals[5] * inv; u1.z = vals[6] * inv; u1.w = vals[7] * inv;
  *(float4*)(p + base)     = u0;
  *(float4*)(p + base + 4) = u1;
}

// ---------------------------------------------------------------------------
// Kernel 4: attn = alpha @ V. Per (b,h): M=2048, N=64, K=2048.
// Output bf16 in [B, S, H*DH] layout. grid (32, 1, 64), block 256.
// ---------------------------------------------------------------------------
__global__ void mha_alpha_v(const float* __restrict__ alpha,
                            const unsigned short* __restrict__ Vh,
                            unsigned short* __restrict__ AttnO) {
  __shared__ unsigned short Alds[64 * 32];   // [q][k]
  __shared__ unsigned short Blds[64 * 32];   // [n=dh][k] (V tile transposed)
  const int q0 = blockIdx.x * 64;
  const int bh = blockIdx.z;
  const int bb = bh >> 4, h = bh & 15;
  const int t  = threadIdx.x;
  const int wave = t >> 5;
  const int wn = (wave & 3) << 4;
  const int wm = (wave >> 2) << 5;
  const int ra = t >> 2, ca = (t & 3) << 3;
  const int rb = t >> 3, cb = (t & 7) << 3;
  const float* arow = alpha + ((size_t)bh * Sc + q0) * Sc;
  const unsigned short* vb = Vh + (size_t)bh * Sc * DHc;
  v8f c0 = {}; v8f c1 = {};

  for (int kk = 0; kk < Sc; kk += 32) {
    {  // stage alpha: f32 -> bf16, vectorized
      const float4* src = (const float4*)(arow + (size_t)ra * Sc + kk + ca);
      const float4 x0 = src[0], x1 = src[1];
      uint4 pk;
      pk.x = pack_bf16x2(x0.x, x0.y); pk.y = pack_bf16x2(x0.z, x0.w);
      pk.z = pack_bf16x2(x1.x, x1.y); pk.w = pack_bf16x2(x1.z, x1.w);
      *(uint4*)(Alds + ra * 32 + ca) = pk;
    }
    {  // stage V transposed: Vh[k][dh] -> Blds[dh][k], bf16 copy (uint4 in)
      union { uint4 q; unsigned short s[8]; } vv;
      vv.q = *(const uint4*)(vb + (size_t)(kk + rb) * DHc + cb);
      unsigned short* bp = Blds + rb;
#pragma unroll
      for (int j = 0; j < 8; ++j) bp[(cb + j) * 32] = vv.s[j];
    }
    if (kk + 32 < Sc) {
      __builtin_prefetch(&arow[(size_t)ra * Sc + kk + 32], 0, 1);
      __builtin_prefetch(&vb[(size_t)(kk + 32 + rb) * DHc], 0, 1);
    }
    __syncthreads();
    const v16bf a0 = load_a_frag(Alds, 32, wm, 0);
    const v16bf a1 = load_a_frag(Alds, 32, wm + 16, 0);
    const v16bf b  = load_b_frag(Blds, 32, wn, 0);
    c0 = WMMA_BF16(a0, b, c0);
    c1 = WMMA_BF16(a1, b, c1);
    __syncthreads();
  }

  const int lane = t & 31;
  const int ln = lane & 15, lh = lane >> 4;
  const int n = wn + ln;                     // dh 0..63
#pragma unroll
  for (int r = 0; r < 8; r += 2) {
    const unsigned p0 = pack_bf16x2(c0[r], c0[r + 1]);
    const unsigned p1 = pack_bf16x2(c1[r], c1[r + 1]);
    {
      const int s = q0 + wm + (lh << 3) + r;
      unsigned short* o = AttnO + ((size_t)bb * Sc + s) * (Hc * DHc) + h * DHc + n;
      o[0] = (unsigned short)p0;
      o[Hc * DHc] = (unsigned short)(p0 >> 16);   // next s -> +1024
    }
    {
      const int s = q0 + wm + 16 + (lh << 3) + r;
      unsigned short* o = AttnO + ((size_t)bb * Sc + s) * (Hc * DHc) + h * DHc + n;
      o[0] = (unsigned short)p1;
      o[Hc * DHc] = (unsigned short)(p1 >> 16);
    }
  }
}

// ---------------------------------------------------------------------------
// Kernel 5: out = attn @ Wo + bo. M=8192, N=1024, K=1024, fp32 output.
// A tile (already bf16) copied with async-to-LDS. grid (128, 16), block 256.
// ---------------------------------------------------------------------------
__global__ void mha_out_proj(const unsigned short* __restrict__ A,  // bf16 [8192,1024]
                             const float* __restrict__ Wo,
                             const float* __restrict__ bo,
                             float* __restrict__ Out) {
  __shared__ unsigned short Alds[64 * 32];
  __shared__ unsigned short Blds[64 * 32];
  const int m0 = blockIdx.x * 64;
  const int n0 = blockIdx.y * 64;
  const int t  = threadIdx.x;
  const int wave = t >> 5;
  const int wn = (wave & 3) << 4;
  const int wm = (wave >> 2) << 5;
  const int ra = t >> 2, ca = (t & 3) << 3;  // A: 16B chunk per thread
  const int rb = t >> 3, cb = (t & 7) << 3;
  const unsigned alds0 = (unsigned)(size_t)(void*)Alds;
  v8f c0 = {}; v8f c1 = {};

  for (int kk = 0; kk < Dc; kk += 32) {
    // stage A: bf16 tile, async memory -> LDS (4KB = 256 x 16B, 1 per thread)
    async_b128(alds0 + (unsigned)(ra * 32 + ca) * 2,
               A + (size_t)(m0 + ra) * 1024 + kk + ca);
    {  // stage Wo transposed: f32 -> bf16
      const float4* ws = (const float4*)(Wo + (size_t)(kk + rb) * Dc + n0 + cb);
      stage_b_transposed(Blds + rb, cb, ws[0], ws[1]);
    }
    if (kk + 32 < Dc) {
      __builtin_prefetch(&A[(size_t)(m0 + ra) * 1024 + kk + 32], 0, 1);
      __builtin_prefetch(&Wo[(size_t)(kk + 32 + rb) * Dc + n0], 0, 1);
    }
    wait_async();
    __syncthreads();
    const v16bf a0 = load_a_frag(Alds, 32, wm, 0);
    const v16bf a1 = load_a_frag(Alds, 32, wm + 16, 0);
    const v16bf b  = load_b_frag(Blds, 32, wn, 0);
    c0 = WMMA_BF16(a0, b, c0);
    c1 = WMMA_BF16(a1, b, c1);
    __syncthreads();
  }

  const int lane = t & 31;
  const int ln = lane & 15, lh = lane >> 4;
  const int n = n0 + wn + ln;
  const float bv = bo[n];
#pragma unroll
  for (int r = 0; r < 8; ++r) {
    Out[(size_t)(m0 + wm + (lh << 3) + r) * Dc + n]      = c0[r] + bv;
    Out[(size_t)(m0 + wm + 16 + (lh << 3) + r) * Dc + n] = c1[r] + bv;
  }
}

// ---------------------------------------------------------------------------
extern "C" void kernel_launch(void* const* d_in, const int* in_sizes, int n_in,
                              void* d_out, int out_size, void* d_ws, size_t ws_size,
                              hipStream_t stream) {
  (void)in_sizes; (void)n_in; (void)out_size; (void)ws_size;
  const float* q  = (const float*)d_in[0];
  const float* k  = (const float*)d_in[1];
  const float* v  = (const float*)d_in[2];
  const float* Wq = (const float*)d_in[3];
  const float* bq = (const float*)d_in[4];
  const float* Wk = (const float*)d_in[5];
  const float* bk = (const float*)d_in[6];
  const float* Wv = (const float*)d_in[7];
  const float* bv = (const float*)d_in[8];
  const float* Wo = (const float*)d_in[9];
  const float* bo = (const float*)d_in[10];

  float* out   = (float*)d_out;                       // [B,S,D] = 8,388,608 f32
  float* alpha = out + (size_t)Bc * Sc * Dc;          // [B,H,S,S] f32

  // Workspace: bf16 Qh/Kh/Vh [B,H,S,DH] + bf16 attn output [B,S,H*DH]. 64 MB.
  unsigned short* Qh    = (unsigned short*)d_ws;
  unsigned short* Kh    = Qh + (size_t)Bc * Hc * Sc * DHc;
  unsigned short* Vh    = Kh + (size_t)Bc * Hc * Sc * DHc;
  unsigned short* AttnO = Vh + (size_t)Bc * Hc * Sc * DHc;

  const dim3 blk(256);
  mha_qkv_proj<<<dim3(128, 16), blk, 0, stream>>>(q, Wq, bq, Qh);
  mha_qkv_proj<<<dim3(128, 16), blk, 0, stream>>>(k, Wk, bk, Kh);
  mha_qkv_proj<<<dim3(128, 16), blk, 0, stream>>>(v, Wv, bv, Vh);
  mha_scores<<<dim3(32, 32, Bc * Hc), blk, 0, stream>>>(Qh, Kh, alpha);
  mha_softmax<<<dim3(Bc * Hc * Sc), blk, 0, stream>>>(alpha);
  mha_alpha_v<<<dim3(32, 1, Bc * Hc), blk, 0, stream>>>(alpha, Vh, AttnO);
  mha_out_proj<<<dim3(128, 16), blk, 0, stream>>>(AttnO, Wo, bo, out);
}
